// selfAttention_24060406792405
// MI455X (gfx1250) — matrix operand order repaired and checked by reading, not compiled, not run on hardware
//
#include <hip/hip_runtime.h>

// CDNA5 / gfx1250: wave32, WMMA 16x16x32 bf16 with f32 accumulate,
// async global->LDS staging (ASYNCcnt) with double buffering.
// V kept transposed [B, D, N] so every LDS fragment read is b128-vectorized.
// Softmax row-sums computed with an extra WMMA against a ones matrix.
typedef __attribute__((ext_vector_type(16))) __bf16 v16bf;
typedef __attribute__((ext_vector_type(8)))  float  v8f;
typedef __attribute__((ext_vector_type(4)))  int    v4i;

#define BB 4
#define NN 4096
#define DD 128

__device__ __forceinline__ v8f wmma_bf16(v16bf a, v16bf b, v8f c) {
  // (neg_a, A, neg_b, B, c_mod, C, reuse_a, reuse_b)
  return __builtin_amdgcn_wmma_f32_16x16x32_bf16(false, a, false, b, (short)0, c,
                                                 false, false);
}

// ---- CDNA5 async global->LDS copy (16B per lane), ASYNCcnt tracked ----
__device__ __forceinline__ void async_copy_b128(void* lds_dst, const void* gsrc) {
#if __has_builtin(__builtin_amdgcn_global_load_async_to_lds_b128)
  __builtin_amdgcn_global_load_async_to_lds_b128(
      (__attribute__((address_space(1))) v4i*)gsrc,
      (__attribute__((address_space(3))) v4i*)lds_dst, 0, 0);
#else
  const unsigned int off = (unsigned int)(unsigned long long)lds_dst;
  asm volatile("global_load_async_to_lds_b128 %0, %1, off"
               :: "v"(off), "v"(gsrc) : "memory");
#endif
}

template <int N>
__device__ __forceinline__ void wait_asynccnt() {
#if __has_builtin(__builtin_amdgcn_s_wait_asynccnt)
  __builtin_amdgcn_s_wait_asynccnt((short)N);
#else
  asm volatile("s_wait_asynccnt %0" :: "i"(N) : "memory");
#endif
}

// A fragment: 16x32 (MxK), row-major source with leading dim `ld`.
// lane m = lane&15; half = lane>>4 selects K-subrange.
// VGPR v : K = 2v + 16*half (lanes 0-15 K=0..15, lanes 16-31 K=16..31)
template <typename T>
__device__ __forceinline__ v16bf load_A_frag(const T* base, int ld) {
  const int lane = threadIdx.x & 31;
  const int m = lane & 15, half = lane >> 4;
  v16bf a;
#pragma unroll
  for (int v = 0; v < 8; ++v) {
    const int kk = 2 * v + 16 * half;
    if constexpr (__is_same(T, __bf16)) {
      a[2 * v]     = base[m * ld + kk];
      a[2 * v + 1] = base[m * ld + kk + 1];
    } else {
      a[2 * v]     = (__bf16)(float)base[m * ld + kk];
      a[2 * v + 1] = (__bf16)(float)base[m * ld + kk + 1];
    }
  }
  return a;
}

// B fragment (32x16, KxN) from a row-major NxK source => B[k][n] = src[n][k].
// lane n = lane&15; lanes 0-15 hold K=0..15, lanes 16-31 K=16..31.
// Per lane: 16 consecutive 16-bit elements -> two ds/global b128 loads.
template <typename T>
__device__ __forceinline__ v16bf load_Bt_frag(const T* base, int ld) {
  const int lane = threadIdx.x & 31;
  const int n = lane & 15, half = lane >> 4;
  v16bf b;
#pragma unroll
  for (int v = 0; v < 8; ++v) {
    const int kk = 2 * v + 16 * half;
    if constexpr (__is_same(T, __bf16)) {
      b[2 * v]     = base[n * ld + kk];
      b[2 * v + 1] = base[n * ld + kk + 1];
    } else {
      b[2 * v]     = (__bf16)(float)base[n * ld + kk];
      b[2 * v + 1] = (__bf16)(float)base[n * ld + kk + 1];
    }
  }
  return b;
}

// OUT = act(A[M,128] @ W[128,128]^T + bias), optional residual / transpose.
// MODE: 0 = plain, 1 = relu, 2 = relu + residual,
//       3 = plain, store transposed bf16 as OUT[B, D, N] (for V).
template <typename TIn, typename TOut, int MODE>
__global__ __launch_bounds__(256) void gemm16(const TIn* __restrict__ A,
                                              const float* __restrict__ W,
                                              const float* __restrict__ bias,
                                              const float* __restrict__ res,
                                              TOut* __restrict__ out) {
  __shared__ __bf16 ldsA[16 * DD];
  const int row0 = blockIdx.x * 16;
  for (int i = threadIdx.x; i < 16 * DD; i += 256)
    ldsA[i] = (__bf16)(float)A[(size_t)row0 * DD + i];
  __syncthreads();

  const int wave = threadIdx.x >> 5;
  const int lane = threadIdx.x & 31;
  const int n1 = lane & 15, half = lane >> 4;
  const int n0 = wave * 16;

  v8f c = {};
#pragma unroll
  for (int kc = 0; kc < 4; ++kc) {
    v16bf a = load_A_frag(ldsA + kc * 32, DD);
    v16bf b = load_Bt_frag(W + (size_t)n0 * DD + kc * 32, DD);
    c = wmma_bf16(a, b, c);
  }
  const float bv = bias[n0 + n1];

  if constexpr (MODE == 3) {
    // Transposed bf16 store: lane covers 8 consecutive tokens at fixed d.
    const int b   = row0 >> 12;              // row0 / NN
    const int tok = (row0 & (NN - 1)) + 8 * half;
    uint4 pkt;
    __bf16* pp = reinterpret_cast<__bf16*>(&pkt);
#pragma unroll
    for (int v = 0; v < 8; ++v) pp[v] = (__bf16)(c[v] + bv);
    *reinterpret_cast<uint4*>(out + ((size_t)b * DD + n0 + n1) * NN + tok) = pkt;
  } else {
#pragma unroll
    for (int v = 0; v < 8; ++v) {
      const size_t idx = (size_t)(row0 + v + 8 * half) * DD + n0 + n1;
      float r = c[v] + bv;
      if constexpr (MODE == 1) r = fmaxf(r, 0.f);
      if constexpr (MODE == 2) r = fmaxf(c[v] + bv, 0.f) + res[idx];
      out[idx] = (TOut)r;
    }
  }
}

// Flash attention: out = relu(softmax(q k^T * scale + eye + adj) @ v) + x.
// Block = 8 waves = 128 query rows of one batch. K (row-major) and V^T
// ([D, N]) streamed 32 keys/step via double-buffered async global->LDS
// copies; online softmax; f32 accumulators; row-sums via WMMA against ones.
__global__ __launch_bounds__(256) void flash_attn(const __bf16* __restrict__ q,
                                                  const __bf16* __restrict__ k,
                                                  const __bf16* __restrict__ vt,
                                                  const float* __restrict__ adj,
                                                  const float* __restrict__ x,
                                                  float* __restrict__ out) {
  __shared__ __bf16 ldsK[2][32 * DD];    // 2 x 8 KB ping-pong, [key][d]
  __shared__ __bf16 ldsVT[2][DD * 32];   // 2 x 8 KB ping-pong, [d][key]
  __shared__ __bf16 ldsP[8][16 * 32];    // 8 KB, wave-private P scratch

  const int batch = blockIdx.x >> 5;               // 32 blocks per batch
  const int wave  = threadIdx.x >> 5;
  const int r0    = ((blockIdx.x & 31) << 7) + (wave << 4);
  const int lane  = threadIdx.x & 31;
  const int n1 = lane & 15, half = lane >> 4;
  const float scale = 0.08838834764831845f;        // 1/sqrt(128)

  const __bf16* kbase  = k  + (size_t)batch * NN * DD;
  const __bf16* vtbase = vt + (size_t)batch * DD * NN;

  // Stage one 32-key chunk: K rows (key-major) + V^T rows (d-major, 64B each).
  // 4 async b128 instructions per wave (16 B per lane each).
  auto issue_chunk = [&](int j0, int buf) {
#pragma unroll
    for (int i = 0; i < 2; ++i) {
      const int idx = threadIdx.x + i * 256;       // 0..511
      async_copy_b128(&ldsK[buf][idx * 8], kbase + (size_t)j0 * DD + idx * 8);
      const int row = idx >> 2, seg = idx & 3;
      async_copy_b128(&ldsVT[buf][row * 32 + seg * 8],
                      vtbase + (size_t)row * NN + j0 + seg * 8);
    }
  };

  // Hold the 16x128 q tile as four 16x32 A fragments for the whole sweep.
  v16bf qf[4];
  const __bf16* qbase = q + ((size_t)batch * NN + r0) * DD;
#pragma unroll
  for (int kc = 0; kc < 4; ++kc) qf[kc] = load_A_frag(qbase + kc * 32, DD);

  // Ones B-matrix for WMMA row-sum reduction of P.
  v16bf onesB;
#pragma unroll
  for (int i = 0; i < 16; ++i) onesB[i] = (__bf16)1.0f;

  v8f acc[8] = {};                                 // 16x128 f32 accumulator
  float mrow[8], lrow[8];
#pragma unroll
  for (int vv = 0; vv < 8; ++vv) { mrow[vv] = -1e30f; lrow[vv] = 0.f; }

  // Per-lane adj base: row (r0 + 8*half + vv), col n1; advances 32 keys/iter.
  const float* adjp = adj + (size_t)(r0 + 8 * half) * NN + n1;

  issue_chunk(0, 0);                               // prologue prefetch

#pragma unroll 1
  for (int j0 = 0; j0 < NN; j0 += 32) {
    const int cur = (j0 >> 5) & 1;
    const bool pf = (j0 + 32) < NN;
    if (pf) issue_chunk(j0 + 32, cur ^ 1);         // prefetch next chunk
    if (pf) wait_asynccnt<4>();                    // current chunk's 4 loads done
    else    wait_asynccnt<0>();
    __syncthreads();

    const __bf16* curK  = ldsK[cur];
    const __bf16* curVT = ldsVT[cur];

    // scores: two 16x16 tiles (keys j0..+15 / j0+16..+31), K=128 in 4 WMMAs each.
    v8f s0 = {}, s1 = {};
#pragma unroll
    for (int kc = 0; kc < 4; ++kc) {
      v16bf b0 = load_Bt_frag(curK + kc * 32, DD);
      s0 = wmma_bf16(qf[kc], b0, s0);
      v16bf b1 = load_Bt_frag(curK + 16 * DD + kc * 32, DD);
      s1 = wmma_bf16(qf[kc], b1, s1);
    }

    // scale + adj bias (+ identity only in the one diagonal chunk),
    // online max update, exp, stash bf16 P into wave-private LDS.
    const bool dchunk = (j0 == (r0 & ~31));        // wave-uniform
    __bf16* pb = ldsP[wave];
    float alpha[8];
#pragma unroll
    for (int vv = 0; vv < 8; ++vv) {
      float a0 = s0[vv] * scale + adjp[vv * NN];
      float a1 = s1[vv] * scale + adjp[vv * NN + 16];
      if (dchunk) {
        const int row = r0 + 8 * half + vv;
        a0 += (row == j0 + n1)      ? 1.f : 0.f;
        a1 += (row == j0 + 16 + n1) ? 1.f : 0.f;
      }

      float t = fmaxf(a0, a1);                     // row-max over this 16-lane half
      t = fmaxf(t, __shfl_xor(t, 1, 32));
      t = fmaxf(t, __shfl_xor(t, 2, 32));
      t = fmaxf(t, __shfl_xor(t, 4, 32));
      t = fmaxf(t, __shfl_xor(t, 8, 32));
      const float mnew = fmaxf(mrow[vv], t);

      alpha[vv] = __expf(mrow[vv] - mnew);
      mrow[vv]  = mnew;
#pragma unroll
      for (int t2 = 0; t2 < 8; ++t2) acc[t2][vv] *= alpha[vv];

      pb[(vv + 8 * half) * 32 + n1]      = (__bf16)__expf(a0 - mnew);
      pb[(vv + 8 * half) * 32 + 16 + n1] = (__bf16)__expf(a1 - mnew);
    }

    // Reload P in A-layout; row-sums via one WMMA against ones (all columns
    // of the result equal the row sum, computed from the same bf16 P as P@V).
    v16bf pa = load_A_frag(pb, 32);
    v8f zc = {};
    v8f rsv = wmma_bf16(pa, onesB, zc);
#pragma unroll
    for (int vv = 0; vv < 8; ++vv)
      lrow[vv] = lrow[vv] * alpha[vv] + rsv[vv];

    // acc += P(16x32) @ V(32x128): 8 column tiles; B from V^T rows (b128 reads).
#pragma unroll
    for (int t = 0; t < 8; ++t) {
      v16bf bvv = load_Bt_frag(curVT + (size_t)t * 16 * 32, 32);
      acc[t] = wmma_bf16(pa, bvv, acc[t]);
    }
    adjp += 32;
    __syncthreads();   // all waves done with `cur` before it is refilled
  }

  // epilogue: out = relu(acc / l) + x
#pragma unroll
  for (int t = 0; t < 8; ++t) {
#pragma unroll
    for (int vv = 0; vv < 8; ++vv) {
      const size_t idx =
          ((size_t)batch * NN + r0 + vv + 8 * half) * DD + t * 16 + n1;
      const float o = acc[t][vv] / lrow[vv];
      out[idx] = fmaxf(o, 0.f) + x[idx];
    }
  }
}

extern "C" void kernel_launch(void* const* d_in, const int* in_sizes, int n_in,
                              void* d_out, int out_size, void* d_ws, size_t ws_size,
                              hipStream_t stream) {
  const float* x   = (const float*)d_in[0];
  const float* adj = (const float*)d_in[1];
  const float* Wq  = (const float*)d_in[2];
  const float* bq  = (const float*)d_in[3];
  const float* Wk  = (const float*)d_in[4];
  const float* bk  = (const float*)d_in[5];
  const float* Wv  = (const float*)d_in[6];
  const float* bv  = (const float*)d_in[7];
  const float* W1  = (const float*)d_in[8];
  const float* b1  = (const float*)d_in[9];
  const float* W2  = (const float*)d_in[10];
  const float* b2  = (const float*)d_in[11];
  float* out = (float*)d_out;

  const size_t E = (size_t)BB * NN * DD;  // 2,097,152 elements
  __bf16* qws  = (__bf16*)d_ws;           // bf16 q  [B, N, D]
  __bf16* kws  = qws + E;                 // bf16 k  [B, N, D]
  __bf16* vtws = kws + E;                 // bf16 v^T [B, D, N]
  float*  attn = (float*)(vtws + E);      // f32 attention output (residual src)
  __bf16* h1 = (__bf16*)(attn + E);       // bf16 hidden
  (void)in_sizes; (void)n_in; (void)out_size; (void)ws_size;

  const int M16 = (BB * NN) / 16;  // 1024 row tiles

  gemm16<float, __bf16, 0><<<M16, 256, 0, stream>>>(x, Wq, bq, nullptr, qws);
  gemm16<float, __bf16, 0><<<M16, 256, 0, stream>>>(x, Wk, bk, nullptr, kws);
  gemm16<float, __bf16, 3><<<M16, 256, 0, stream>>>(x, Wv, bv, nullptr, vtws);

  flash_attn<<<(BB * NN) / 128, 256, 0, stream>>>(qws, kws, vtws, adj, x, attn);

  gemm16<float, __bf16, 1><<<M16, 256, 0, stream>>>(attn, W1, b1, nullptr, h1);
  gemm16<__bf16, float, 2><<<M16, 256, 0, stream>>>(h1, W2, b2, attn, out);
}
